// MyGCN_70798240907405
// MI455X (gfx1250) — matrix operand order repaired and checked by reading, not compiled, not run on hardware
//
#include <hip/hip_runtime.h>
#include <hip/hip_bf16.h>

typedef __bf16 bf16_t;
typedef __attribute__((ext_vector_type(16))) __bf16 v16bf;
typedef __attribute__((ext_vector_type(8)))  __bf16 v8bf;
typedef __attribute__((ext_vector_type(8)))  float  v8f;

__device__ __forceinline__ unsigned short f32_to_bf16(float f) {
    unsigned int u = __float_as_uint(f);
    u += 0x7FFFu + ((u >> 16) & 1u);   // round-to-nearest-even
    return (unsigned short)(u >> 16);
}

// ---------------- degree / dinv ----------------
__global__ void k_deg_init(float* deg, int n) {
    int i = blockIdx.x * blockDim.x + threadIdx.x;
    if (i < n) deg[i] = 1.0f;                       // self loop
}
__global__ void k_deg_accum(const int* __restrict__ dst, float* deg, int e) {
    int i = blockIdx.x * blockDim.x + threadIdx.x;
    if (i < e) atomicAdd(&deg[dst[i]], 1.0f);
}
__global__ void k_dinv(float* d, int n) {
    int i = blockIdx.x * blockDim.x + threadIdx.x;
    if (i < n) { float v = d[i]; d[i] = (v > 0.f) ? rsqrtf(v) : 0.f; }
}

// ---------------- casts ----------------
__global__ void k_cast_bf16(const float* __restrict__ in, unsigned short* __restrict__ out,
                            long long total) {
    long long i = (long long)blockIdx.x * blockDim.x + threadIdx.x;
    if (i < total) out[i] = f32_to_bf16(in[i]);
}
// W is K x C row-major (K=256). Produce Wt: C x K (column-major B) in bf16.
__global__ void k_transpose_cast_W(const float* __restrict__ W, unsigned short* __restrict__ Wt,
                                   int C) {
    int i = blockIdx.x * blockDim.x + threadIdx.x;   // over 256*C
    if (i >= 256 * C) return;
    int k = i / C, c = i % C;
    Wt[c * 256 + k] = f32_to_bf16(W[i]);
}

// ---------------- WMMA GEMM: C[M x Cout] = A[M x 256](bf16) * B(from Wt: Cout x 256 bf16) ----------------
// One wave computes one 16x16 tile. Block = 256 threads = 8 waves = 8 column tiles (128 cols).
__global__ void k_gemm_wmma(const unsigned short* __restrict__ Abf,
                            const unsigned short* __restrict__ Wt,
                            float* __restrict__ C, int M, int Cout) {
    constexpr int K = 256;
    const int lane  = threadIdx.x & 31;
    const int wave  = threadIdx.x >> 5;
    const int tileM = blockIdx.x;
    const int tileN = blockIdx.y * 8 + wave;
    if (tileN * 16 >= Cout) return;                  // uniform per wave

    int row = tileM * 16 + (lane & 15);
    int rclamp = row < M ? row : (M - 1);            // keep EXEC all-ones for WMMA
    const int col = tileN * 16 + (lane & 15);
    const int kkA = (lane >> 4) * 8;                 // A: hi lanes start at K+8 per ISA layout
    const int kkB = (lane >> 4) * 8;                 // B: same K layout as A

    const bf16_t* arow = (const bf16_t*)Abf + (size_t)rclamp * K;
    const bf16_t* bcol = (const bf16_t*)Wt  + (size_t)col * K;

    v8f acc = {0.f, 0.f, 0.f, 0.f, 0.f, 0.f, 0.f, 0.f};
#pragma unroll
    for (int k0 = 0; k0 < K; k0 += 32) {
        union { v16bf v; v8bf h[2]; } a;
        a.h[0] = *(const v8bf*)(arow + k0 + kkA);        // K = k0+kk .. +7
        a.h[1] = *(const v8bf*)(arow + k0 + 16 + kkA);   // K = k0+16+kk .. +7
        union { v16bf v; v8bf h[2]; } b;
        b.h[0] = *(const v8bf*)(bcol + k0 + kkB);        // K = k0+kk .. +7
        b.h[1] = *(const v8bf*)(bcol + k0 + 16 + kkB);   // K = k0+16+kk .. +7
        acc = __builtin_amdgcn_wmma_f32_16x16x32_bf16(
            /*neg_a=*/false, a.v, /*neg_b=*/false, b.v,
            /*c_mod=*/(short)0, acc, /*reuse_a=*/false, /*reuse_b=*/false);
    }
    // C/D layout: VGPR r -> M = tileM*16 + r + 8*(lane>=16), N = tileN*16 + lane%16
    const int m0 = tileM * 16 + (lane >> 4) * 8;
    const int n  = tileN * 16 + (lane & 15);
#pragma unroll
    for (int r = 0; r < 8; ++r) {
        int m = m0 + r;
        if (m < M) C[(size_t)m * Cout + n] = acc[r];
    }
}

// ---------------- aggregation ----------------
// out[v][c] = h[v][c] * dinv[v]^2   (self-loop term; also initializes the buffer)
__global__ void k_selfloop(const float* __restrict__ h, const float* __restrict__ dinv,
                           float* __restrict__ out, int n, int C) {
    long long i = (long long)blockIdx.x * blockDim.x + threadIdx.x;
    if (i >= (long long)n * C) return;
    int v = (int)(i / C);
    float di = dinv[v];
    out[i] = h[i] * di * di;
}
// per edge: out[dst] += h[src] * dinv[src]*dinv[dst]; one wave per edge
__global__ void k_agg_edges(const float* __restrict__ h, const float* __restrict__ dinv,
                            const int* __restrict__ src, const int* __restrict__ dst,
                            float* __restrict__ out, int e, int C) {
    int lane = threadIdx.x & 31;
    int ei = blockIdx.x * (blockDim.x >> 5) + (threadIdx.x >> 5);
    if (ei >= e) return;
    int s = src[ei], d = dst[ei];
    float norm = dinv[s] * dinv[d];
    const float* hs = h + (size_t)s * C;
    float* od = out + (size_t)d * C;
    for (int c = lane; c < C; c += 32)
        atomicAdd(&od[c], hs[c] * norm);
}
// layers 1,2: x_next_bf16 = relu(out + b)
__global__ void k_bias_relu_cast(const float* __restrict__ agg, const float* __restrict__ bias,
                                 unsigned short* __restrict__ nxt, int n, int C) {
    long long i = (long long)blockIdx.x * blockDim.x + threadIdx.x;
    if (i >= (long long)n * C) return;
    int c = (int)(i % C);
    float v = agg[i] + bias[c];
    v = fmaxf(v, 0.f);
    nxt[i] = f32_to_bf16(v);
}
// layer 3: agg += b (in place, f32, no relu)
__global__ void k_bias_f32(float* __restrict__ agg, const float* __restrict__ bias, int n, int C) {
    long long i = (long long)blockIdx.x * blockDim.x + threadIdx.x;
    if (i >= (long long)n * C) return;
    agg[i] += bias[(int)(i % C)];
}

// ---------------- pooling ----------------
__global__ void k_pool_accum(const float* __restrict__ h, const int* __restrict__ batch,
                             float* __restrict__ pool, float* __restrict__ cnt, int n, int C) {
    int lane = threadIdx.x & 31;
    int v = blockIdx.x * (blockDim.x >> 5) + (threadIdx.x >> 5);
    if (v >= n) return;
    int g = batch[v];
    const float* hv = h + (size_t)v * C;
    float* pg = pool + (size_t)g * C;
    for (int c = lane; c < C; c += 32) atomicAdd(&pg[c], hv[c]);
    if (lane == 0) atomicAdd(&cnt[g], 1.0f);
}
__global__ void k_pool_div(const float* __restrict__ pool, const float* __restrict__ cnt,
                           float* __restrict__ out, int C) {
    int g = blockIdx.x, c = threadIdx.x;
    out[g * C + c] = pool[g * C + c] / fmaxf(cnt[g], 1.0f);
}

extern "C" void kernel_launch(void* const* d_in, const int* in_sizes, int n_in,
                              void* d_out, int out_size, void* d_ws, size_t ws_size,
                              hipStream_t stream) {
    const float* x  = (const float*)d_in[0];
    const int*   ei = (const int*)d_in[1];
    const int*   batch = (const int*)d_in[2];
    const float* Wl[3] = { (const float*)d_in[3], (const float*)d_in[5], (const float*)d_in[7] };
    const float* bl[3] = { (const float*)d_in[4], (const float*)d_in[6], (const float*)d_in[8] };
    float* out = (float*)d_out;

    const int n = in_sizes[0] / 256;      // 100000
    const int e = in_sizes[1] / 2;        // 3200000
    const int* src = ei;
    const int* dst = ei + e;
    const int NG = 64, OUTC = 128;

    // ---- workspace layout (256B aligned slabs) ----
    char* ws = (char*)d_ws;
    size_t cur = 0;
    auto carve = [&](size_t bytes) { size_t o = cur; cur += (bytes + 255) & ~(size_t)255; return o; };
    unsigned short* xb   = (unsigned short*)(ws + carve((size_t)n * 256 * 2)); // bf16 layer input
    float*          hge  = (float*)(ws + carve((size_t)n * 256 * 4));          // GEMM output
    float*          hag  = (float*)(ws + carve((size_t)n * 256 * 4));          // aggregated
    float*          dinv = (float*)(ws + carve((size_t)n * 4));
    unsigned short* Wt   = (unsigned short*)(ws + carve((size_t)256 * 256 * 2));
    float*          pool = (float*)(ws + carve((size_t)(NG * OUTC + NG) * 4));
    float*          cnt  = pool + NG * OUTC;
    (void)ws_size; (void)n_in; (void)out_size;

    const int TB = 256;
    // degree -> dinv (reuse dinv buffer for degree)
    k_deg_init <<<(n + TB - 1) / TB, TB, 0, stream>>>(dinv, n);
    k_deg_accum<<<(e + TB - 1) / TB, TB, 0, stream>>>(dst, dinv, e);
    k_dinv     <<<(n + TB - 1) / TB, TB, 0, stream>>>(dinv, n);

    // cast x -> bf16
    long long nx = (long long)n * 256;
    k_cast_bf16<<<(unsigned)((nx + TB - 1) / TB), TB, 0, stream>>>(x, xb, nx);

    const int cout[3]  = {256, 256, 128};
    for (int L = 0; L < 3; ++L) {
        const int C = cout[L];
        const long long nc = (long long)n * C;
        // B^T in bf16
        k_transpose_cast_W<<<(256 * C + TB - 1) / TB, TB, 0, stream>>>(Wl[L], Wt, C);
        // h = x @ W  (WMMA)
        dim3 gg((n + 15) / 16, (C + 127) / 128);
        k_gemm_wmma<<<gg, 256, 0, stream>>>(xb, Wt, hge, n, C);
        // aggregation: self-loop init, then edge scatter
        k_selfloop <<<(unsigned)((nc + TB - 1) / TB), TB, 0, stream>>>(hge, dinv, hag, n, C);
        k_agg_edges<<<(e + 7) / 8, 256, 0, stream>>>(hge, dinv, src, dst, hag, e, C);
        if (L < 2)
            k_bias_relu_cast<<<(unsigned)((nc + TB - 1) / TB), TB, 0, stream>>>(hag, bl[L], xb, n, C);
        else
            k_bias_f32<<<(unsigned)((nc + TB - 1) / TB), TB, 0, stream>>>(hag, bl[L], n, C);
    }

    // global mean pool
    hipMemsetAsync(pool, 0, (size_t)(NG * OUTC + NG) * sizeof(float), stream);
    k_pool_accum<<<(n + 7) / 8, 256, 0, stream>>>(hag, batch, pool, cnt, n, OUTC);
    k_pool_div  <<<NG, OUTC, 0, stream>>>(pool, cnt, out, OUTC);
}
